// LocalAttentionBlock_5660766896390
// MI455X (gfx1250) — matrix operand order repaired and checked by reading, not compiled, not run on hardware
//
#include <hip/hip_runtime.h>

// ---------------------------------------------------------------------------
// Problem constants (match reference)
// ---------------------------------------------------------------------------
#define T_  8192
#define D_  1024
#define H_  16
#define DH_ 64
#define WB_ 128
#define NB_ 64
#define WIN_ 128
#define SLD 392          // fp32 score-row stride (384 cols + pad, multiple of 8)

typedef __bf16 bhalf;
typedef __attribute__((ext_vector_type(16))) __bf16 v16bf;
typedef __attribute__((ext_vector_type(8)))  float  v8f;
typedef unsigned int u32x4 __attribute__((ext_vector_type(4)));

union FragU { v16bf v; u32x4 q[2]; };

// ---------------------------------------------------------------------------
// bf16 helpers (round-to-nearest-even)
// ---------------------------------------------------------------------------
__device__ __forceinline__ bhalf f2bf(float f) {
  union { float f; unsigned u; } a; a.f = f;
  unsigned r = a.u + 0x7FFFu + ((a.u >> 16) & 1u);
  unsigned short h = (unsigned short)(r >> 16);
  bhalf b; __builtin_memcpy(&b, &h, 2);
  return b;
}
__device__ __forceinline__ float bf2f(bhalf b) {
  unsigned short h; __builtin_memcpy(&h, &b, 2);
  union { unsigned u; float f; } a; a.u = ((unsigned)h) << 16;
  return a.f;
}
__device__ __forceinline__ int iclamp(int v, int lo, int hi) {
  return v < lo ? lo : (v > hi ? hi : v);
}

// ---------------------------------------------------------------------------
// WMMA fragment loaders (wave32; contiguous k order, same for A and B)
// A 16x32 bf16: lanes 0-15 -> M=lane, K 0..15; lanes 16-31 -> K 16..31
// ---------------------------------------------------------------------------
__device__ __forceinline__ v16bf frag_a(const bhalf* __restrict__ base, int ld) {
  const int lane = threadIdx.x & 31;
  const int r  = lane & 15;
  const int kh = (lane >> 4) * 16;
  const bhalf* p = base + (size_t)r * ld + kh;
  FragU t;
  t.q[0] = *(const u32x4*)(p);        // K = kh .. kh+7
  t.q[1] = *(const u32x4*)(p + 8);    // K = kh+8 .. kh+15
  return t.v;
}
// B 32x16 bf16 from an N-major [N][K] buffer: lane%16 = N, 16 contiguous K,
// lanes 0-15 -> K 0..15, lanes 16-31 -> K 16..31
__device__ __forceinline__ v16bf frag_b(const bhalf* __restrict__ base, int ld) {
  const int lane = threadIdx.x & 31;
  const int n  = lane & 15;
  const int k0 = (lane >> 4) * 16;
  const bhalf* p = base + (size_t)n * ld + k0;
  FragU t;
  t.q[0] = *(const u32x4*)(p);
  t.q[1] = *(const u32x4*)(p + 8);
  return t.v;
}
__device__ __forceinline__ v8f wmma_bf16(v16bf a, v16bf b, v8f c) {
  return __builtin_amdgcn_wmma_f32_16x16x32_bf16(false, a, false, b, (short)0, c,
                                                 false, false);
}

// ---------------------------------------------------------------------------
// Weight transpose + fp32 -> bf16 convert:  in [R][C] fp32  ->  out [C][R] bf16
// ---------------------------------------------------------------------------
__global__ void cvt_transpose(const float* __restrict__ in, bhalf* __restrict__ out,
                              int R, int C) {
  size_t i = (size_t)blockIdx.x * 256 + threadIdx.x;
  if (i >= (size_t)R * C) return;
  int r = (int)(i / C), c = (int)(i % C);
  out[(size_t)c * R + r] = f2bf(in[i]);
}

// ---------------------------------------------------------------------------
// LayerNorm (one row per block), fp32 in -> bf16 out
// ---------------------------------------------------------------------------
__global__ __launch_bounds__(256) void layernorm_bf16(
    const float* __restrict__ x, const float* __restrict__ g,
    const float* __restrict__ b, bhalf* __restrict__ out) {
  const int row = blockIdx.x;
  const float* xr = x + (size_t)row * D_;
  float vals[4], s = 0.f, s2 = 0.f;
#pragma unroll
  for (int i = 0; i < 4; ++i) {
    float v = xr[threadIdx.x + i * 256];
    vals[i] = v; s += v; s2 += v * v;
  }
#pragma unroll
  for (int off = 16; off; off >>= 1) {
    s  += __shfl_xor(s,  off, 32);
    s2 += __shfl_xor(s2, off, 32);
  }
  __shared__ float red[2][8];
  const int wave = threadIdx.x >> 5, lane = threadIdx.x & 31;
  if (lane == 0) { red[0][wave] = s; red[1][wave] = s2; }
  __syncthreads();
  s = 0.f; s2 = 0.f;
#pragma unroll
  for (int w = 0; w < 8; ++w) { s += red[0][w]; s2 += red[1][w]; }
  const float mu  = s  * (1.f / D_);
  const float var = s2 * (1.f / D_) - mu * mu;
  const float rs  = rsqrtf(var + 1e-5f);
#pragma unroll
  for (int i = 0; i < 4; ++i) {
    int c = threadIdx.x + i * 256;
    out[(size_t)row * D_ + c] = f2bf((vals[i] - mu) * rs * g[c] + b[c]);
  }
}

// ---------------------------------------------------------------------------
// WMMA GEMM: C[M,N] = A[M,K](bf16,row-major) x Bt[N,K](bf16,N-major)
// 128x128 tile per block, BK=64 stage, register-prefetch pipeline.
// 8 waves, each wave 64x32 (4x2 WMMA tiles, 16 WMMAs per stage).
// MODE 0: store bf16            MODE 1: fp32 out = acc + res
// MODE 2: silu(acc) -> bf16     MODE 3: bf16 store transposed (out[N][M])
// ---------------------------------------------------------------------------
template <int MODE>
__global__ __launch_bounds__(256) void gemm_bf16_128(
    const bhalf* __restrict__ A, const bhalf* __restrict__ Bt,
    const float* __restrict__ res, void* __restrict__ out,
    int M, int N, int K) {
  __shared__ bhalf As[128 * 64];
  __shared__ bhalf Bs[128 * 64];
  const int tid  = threadIdx.x;
  const int lane = tid & 31;
  const int wave = tid >> 5;
  const int m0 = blockIdx.y * 128;
  const int n0 = blockIdx.x * 128;
  const int wm = (wave >> 2) * 64;   // wave row offset within tile
  const int wn = (wave & 3) * 32;    // wave col offset within tile

  v8f acc[4][2];
#pragma unroll
  for (int i = 0; i < 4; ++i)
#pragma unroll
    for (int j = 0; j < 2; ++j) acc[i][j] = v8f{0, 0, 0, 0, 0, 0, 0, 0};

  // each thread stages one 64-byte chunk per matrix: row = tid>>1, 32 cols
  const int lrow = tid >> 1;
  const int lcol = (tid & 1) * 32;
  const bhalf* gA = A  + (size_t)(m0 + lrow) * K + lcol;
  const bhalf* gB = Bt + (size_t)(n0 + lrow) * K + lcol;
  u32x4* la = (u32x4*)(As + lrow * 64 + lcol);
  u32x4* lb = (u32x4*)(Bs + lrow * 64 + lcol);

  u32x4 pa[4], pb[4];
#pragma unroll
  for (int u = 0; u < 4; ++u) {
    pa[u] = ((const u32x4*)gA)[u];
    pb[u] = ((const u32x4*)gB)[u];
  }

  for (int kt = 0; kt < K; kt += 64) {
#pragma unroll
    for (int u = 0; u < 4; ++u) { la[u] = pa[u]; lb[u] = pb[u]; }
    __syncthreads();
    if (kt + 64 < K) {  // prefetch next stage while WMMAs run
      const u32x4* na = (const u32x4*)(gA + kt + 64);
      const u32x4* nb = (const u32x4*)(gB + kt + 64);
#pragma unroll
      for (int u = 0; u < 4; ++u) { pa[u] = na[u]; pb[u] = nb[u]; }
    }
#pragma unroll
    for (int ks = 0; ks < 64; ks += 32) {
      v16bf af[4], bfr[2];
#pragma unroll
      for (int i = 0; i < 4; ++i) af[i]  = frag_a(As + (wm + i * 16) * 64 + ks, 64);
#pragma unroll
      for (int j = 0; j < 2; ++j) bfr[j] = frag_b(Bs + (wn + j * 16) * 64 + ks, 64);
#pragma unroll
      for (int i = 0; i < 4; ++i)
#pragma unroll
        for (int j = 0; j < 2; ++j)
          acc[i][j] = wmma_bf16(af[i], bfr[j], acc[i][j]);
    }
    __syncthreads();
  }

  const int cl = lane & 15;
  const int rh = (lane >> 4) * 8;
#pragma unroll
  for (int i = 0; i < 4; ++i)
#pragma unroll
    for (int j = 0; j < 2; ++j)
#pragma unroll
      for (int e = 0; e < 8; ++e) {
        const int r = m0 + wm + i * 16 + rh + e;
        const int c = n0 + wn + j * 16 + cl;
        const float v = acc[i][j][e];
        const size_t idx = (size_t)r * N + c;
        if constexpr (MODE == 0) {
          ((bhalf*)out)[idx] = f2bf(v);
        } else if constexpr (MODE == 1) {
          ((float*)out)[idx] = v + res[idx];
        } else if constexpr (MODE == 2) {
          const float sg = __builtin_amdgcn_rcpf(1.f + __expf(-v));
          ((bhalf*)out)[idx] = f2bf(v * sg);
        } else {
          ((bhalf*)out)[(size_t)c * M + r] = f2bf(v);
        }
      }
}

// ---------------------------------------------------------------------------
// Block-local banded attention. One wave (32 threads) per 16 query rows.
// grid = (8 strips, NB, H). q,k: bf16 [T][D]; vT: bf16 [D][T]; out: bf16 [T][D]
// ---------------------------------------------------------------------------
__global__ __launch_bounds__(32) void local_attn(
    const bhalf* __restrict__ q, const bhalf* __restrict__ k,
    const bhalf* __restrict__ vT, bhalf* __restrict__ out) {
  __shared__ float sc[16 * SLD];              // 25 KB fp32 scores, reused as bf16 probs
  const int lane = threadIdx.x;
  const int head = blockIdx.z;
  const int nb   = blockIdx.y;
  const int q0   = nb * WB_ + blockIdx.x * 16;  // first query row of this wave
  const int hcol = head * DH_;
  const int cl   = lane & 15;
  const int rh   = (lane >> 4) * 8;

  // Q fragments (16 rows x 64 dh) straight from global, reused for all key tiles
  const v16bf a0 = frag_a(q + (size_t)q0 * D_ + hcol,      D_);
  const v16bf a1 = frag_a(q + (size_t)q0 * D_ + hcol + 32, D_);

  // ---- phase 1: scores = Q K^T * 1/8, masked, into LDS fp32 ----
  for (int nt = 0; nt < 24; ++nt) {            // 24 key tiles of 16 over [prev|self|next]
    const int key_base = nb * WB_ - WB_ + nt * 16;
    v8f acc = v8f{0, 0, 0, 0, 0, 0, 0, 0};
#pragma unroll
    for (int ks = 0; ks < 2; ++ks) {
      // B fragment: lane%16 = key column, 16 contiguous dh values (clamped row)
      const int key = iclamp(key_base + cl, 0, T_ - 1);
      const int k0  = (lane >> 4) * 16;
      const bhalf* p = k + (size_t)key * D_ + hcol + ks * 32 + k0;
      FragU t;
      t.q[0] = *(const u32x4*)(p);
      t.q[1] = *(const u32x4*)(p + 8);
      acc = wmma_bf16(ks == 0 ? a0 : a1, t.v, acc);
    }
    const int key_pos = key_base + cl;
#pragma unroll
    for (int e = 0; e < 8; ++e) {
      const int qi  = q0 + rh + e;
      const int rel = key_pos - qi;
      const bool ok = (rel >= -WIN_) && (rel <= WIN_) && (key_pos >= 0) && (key_pos < T_);
      sc[(rh + e) * SLD + nt * 16 + cl] = ok ? acc[e] * 0.125f : -1e30f;
    }
  }
  __syncthreads();

  // ---- phase 2: row softmax, fp32 -> bf16 in place (lane handles one row) ----
  if (lane < 16) {
    float* row = sc + lane * SLD;
    float m = -1e30f;
    for (int c = 0; c < 384; ++c) m = fmaxf(m, row[c]);
    bhalf* prow = (bhalf*)row;                 // bf16 col c at byte 2c (< fp32 byte 4c: safe)
    float s = 0.f;
    for (int c = 0; c < 384; ++c) {
      const float e = __expf(row[c] - m);
      s += e;
      prow[c] = f2bf(e);
    }
    const float inv = 1.f / s;
    for (int c = 0; c < 384; ++c) prow[c] = f2bf(bf2f(prow[c]) * inv);
  }
  __syncthreads();

  // ---- phase 3: out = P V  (P: 16x384 bf16 in LDS, ld = 2*SLD) ----
  const bhalf* probs = (const bhalf*)sc;
#pragma unroll
  for (int dt = 0; dt < 4; ++dt) {
    v8f acc = v8f{0, 0, 0, 0, 0, 0, 0, 0};
    for (int ks = 0; ks < 12; ++ks) {
      const v16bf a = frag_a(probs + ks * 32, 2 * SLD);
      // B fragment from vT[D][T]: lane%16 = dh col, 16 contiguous keys (clamped)
      const int dh = hcol + dt * 16 + cl;
      const int kk = iclamp(nb * WB_ - WB_ + ks * 32 + (lane >> 4) * 16, 0, T_ - 16);
      const bhalf* p = vT + (size_t)dh * T_ + kk;
      FragU t;
      t.q[0] = *(const u32x4*)(p);
      t.q[1] = *(const u32x4*)(p + 8);
      acc = wmma_bf16(a, t.v, acc);
    }
#pragma unroll
    for (int e = 0; e < 8; ++e)
      out[(size_t)(q0 + rh + e) * D_ + hcol + dt * 16 + cl] = f2bf(acc[e]);
  }
}

// ---------------------------------------------------------------------------
// Host-side orchestration
// ---------------------------------------------------------------------------
extern "C" void kernel_launch(void* const* d_in, const int* in_sizes, int n_in,
                              void* d_out, int out_size, void* d_ws, size_t ws_size,
                              hipStream_t stream) {
  const float* x      = (const float*)d_in[0];
  const float* wq     = (const float*)d_in[1];
  const float* wk     = (const float*)d_in[2];
  const float* wv     = (const float*)d_in[3];
  const float* wo     = (const float*)d_in[4];
  const float* wg     = (const float*)d_in[5];
  const float* wd     = (const float*)d_in[6];
  const float* norm_g = (const float*)d_in[7];
  const float* norm_b = (const float*)d_in[8];
  const float* ffn_g  = (const float*)d_in[9];
  const float* ffn_b  = (const float*)d_in[10];
  float* out = (float*)d_out;

  char* w = (char*)d_ws;
  auto take = [&](size_t bytes) -> char* {
    char* p = w;
    w += (bytes + 255) & ~(size_t)255;
    return p;
  };
  bhalf* hb   = (bhalf*)take((size_t)T_ * D_ * 2);
  bhalf* qb   = (bhalf*)take((size_t)T_ * D_ * 2);
  bhalf* kb   = (bhalf*)take((size_t)T_ * D_ * 2);
  bhalf* vT   = (bhalf*)take((size_t)D_ * T_ * 2);
  bhalf* attn = (bhalf*)take((size_t)T_ * D_ * 2);
  float* x1   = (float*)take((size_t)T_ * D_ * 4);
  bhalf* h2   = (bhalf*)take((size_t)T_ * D_ * 2);
  bhalf* gbuf = (bhalf*)take((size_t)T_ * 2 * D_ * 2);
  bhalf* wqT  = (bhalf*)take((size_t)D_ * D_ * 2);
  bhalf* wkT  = (bhalf*)take((size_t)D_ * D_ * 2);
  bhalf* wvT  = (bhalf*)take((size_t)D_ * D_ * 2);
  bhalf* woT  = (bhalf*)take((size_t)D_ * D_ * 2);
  bhalf* wgT  = (bhalf*)take((size_t)D_ * 2 * D_ * 2);
  bhalf* wdT  = (bhalf*)take((size_t)2 * D_ * D_ * 2);
  (void)ws_size; (void)in_sizes; (void)n_in; (void)out_size;

  const int nsq = (D_ * D_ + 255) / 256;
  const int ndb = (2 * D_ * D_ + 255) / 256;
  cvt_transpose<<<nsq, 256, 0, stream>>>(wq, wqT, D_, D_);
  cvt_transpose<<<nsq, 256, 0, stream>>>(wk, wkT, D_, D_);
  cvt_transpose<<<nsq, 256, 0, stream>>>(wv, wvT, D_, D_);
  cvt_transpose<<<nsq, 256, 0, stream>>>(wo, woT, D_, D_);
  cvt_transpose<<<ndb, 256, 0, stream>>>(wg, wgT, D_, 2 * D_);
  cvt_transpose<<<ndb, 256, 0, stream>>>(wd, wdT, 2 * D_, D_);

  layernorm_bf16<<<T_, 256, 0, stream>>>(x, norm_g, norm_b, hb);

  const dim3 gsq(D_ / 128, T_ / 128);
  gemm_bf16_128<0><<<gsq, 256, 0, stream>>>(hb, wqT, nullptr, qb, T_, D_, D_);
  gemm_bf16_128<0><<<gsq, 256, 0, stream>>>(hb, wkT, nullptr, kb, T_, D_, D_);
  gemm_bf16_128<3><<<gsq, 256, 0, stream>>>(hb, wvT, nullptr, vT, T_, D_, D_);

  local_attn<<<dim3(WB_ / 16, NB_, H_), 32, 0, stream>>>(qb, kb, vT, attn);

  gemm_bf16_128<1><<<gsq, 256, 0, stream>>>(attn, woT, x, x1, T_, D_, D_);

  layernorm_bf16<<<T_, 256, 0, stream>>>(x1, ffn_g, ffn_b, h2);

  gemm_bf16_128<2><<<dim3(2 * D_ / 128, T_ / 128), 256, 0, stream>>>(
      h2, wgT, nullptr, gbuf, T_, 2 * D_, D_);
  gemm_bf16_128<1><<<gsq, 256, 0, stream>>>(gbuf, wdT, x1, out, T_, D_, 2 * D_);
}
